// GAT_43843026157853
// MI455X (gfx1250) — matrix operand order repaired and checked by reading, not compiled, not run on hardware
//
#include <hip/hip_runtime.h>

// ---------------- problem constants ----------------
#define NN     50000          // nodes
#define EE     800000         // edges
#define EP     (EE + NN)      // edges + self loops
#define FIN    500            // input features
#define KP1    512            // FIN padded to multiple of 32
#define HEADS  8
#define HID    16
#define C1     128            // HEADS*HID
#define NCLS   40
#define C2P    48             // NCLS padded to multiple of 16
#define NEG_SLOPE 0.2f

typedef _Float16 v8h  __attribute__((ext_vector_type(8)));
typedef _Float16 v16h __attribute__((ext_vector_type(16)));
typedef float    v8f  __attribute__((ext_vector_type(8)));

// float atomic max via sign-magnitude trick (init must be -inf = 0xFF800000)
__device__ __forceinline__ void atomicMaxF(float* addr, float v) {
  if (v >= 0.0f) atomicMax((int*)addr, __float_as_int(v));
  else           atomicMin((unsigned int*)addr, __float_as_uint(v));
}

__device__ __forceinline__ void edge_sd(const int* __restrict__ ei, int e,
                                        int& s, int& d) {
  if (e < EE) { s = ei[e]; d = ei[EE + e]; }
  else        { s = d = e - EE; }             // self loop
}

__device__ __forceinline__ v16h cat16(v8h lo, v8h hi) {
  return __builtin_shufflevector(lo, hi, 0,1,2,3,4,5,6,7,8,9,10,11,12,13,14,15);
}

// ---------------- utility fills ----------------
__global__ void fill_u32_kernel(unsigned* __restrict__ p, long long n, unsigned v) {
  long long i = (long long)blockIdx.x * blockDim.x + threadIdx.x;
  if (i < n) p[i] = v;
}

// ---------------- f16 conversion / transposition ----------------
__global__ void cvt_x_kernel(const float* __restrict__ x, _Float16* __restrict__ xh) {
  long long i = (long long)blockIdx.x * blockDim.x + threadIdx.x;
  if (i >= (long long)NN * KP1) return;
  int n = (int)(i >> 9), k = (int)(i & 511);
  xh[i] = (k < FIN) ? (_Float16)x[(long long)n * FIN + k] : (_Float16)0.0f;
}

__global__ void cvt_w1_kernel(const float* __restrict__ W1, _Float16* __restrict__ w1t) {
  int i = blockIdx.x * blockDim.x + threadIdx.x;
  if (i >= C1 * KP1) return;
  int c = i >> 9, k = i & 511;                 // w1t[c][k] = W1[k][c]
  w1t[i] = (k < FIN) ? (_Float16)W1[k * C1 + c] : (_Float16)0.0f;
}

__global__ void cvt_w2_kernel(const float* __restrict__ W2, _Float16* __restrict__ w2t) {
  int i = blockIdx.x * blockDim.x + threadIdx.x;
  if (i >= C2P * C1) return;
  int c = i >> 7, k = i & 127;                 // w2t[c][k] = W2[k][c], zero-pad cols
  w2t[i] = (c < NCLS) ? (_Float16)W2[k * NCLS + c] : (_Float16)0.0f;
}

// ---------------- GEMM1: h1[50000x128] = xh[50000x512] @ w1t^T, WMMA f16 ----------------
// Block = 128 threads (4 waves). Block computes 16 rows x 128 cols.
// A tile (16x512 f16, 16KB) staged once in LDS, shared by all 4 waves.
// Each wave computes a 16x32 tile: one A fragment feeds two WMMAs (2 accumulators).
__global__ __launch_bounds__(128) void gemm1_wmma_kernel(
    const _Float16* __restrict__ A, const _Float16* __restrict__ Bt,
    float* __restrict__ D) {
  __shared__ _Float16 aTile[16 * KP1];              // 16 KB of the 320KB/WGP LDS
  const int tid  = threadIdx.x;
  const int lane = tid & 31;
  const int wave = tid >> 5;                        // 0..3
  const int t    = lane & 15;
  const int sel  = lane >> 4;                       // 0: lanes 0-15, 1: lanes 16-31
  const int rb   = blockIdx.x << 4;

  // cooperative stage: 16 rows x 512 halves = 1024 x b128, 8 per thread
  {
    const v8h* src = (const v8h*)(A + (size_t)rb * KP1);
    v8h* dst = (v8h*)aTile;
#pragma unroll
    for (int i = 0; i < 8; ++i) dst[tid + 128 * i] = src[tid + 128 * i];
  }
  __syncthreads();

  const int cb = wave << 5;                         // 32 cols per wave
  const _Float16* arow  = aTile + (size_t)t * KP1;  // LDS fragment source
  const _Float16* bc0   = Bt + (size_t)(cb + t) * KP1;
  const _Float16* bc1   = Bt + (size_t)(cb + 16 + t) * KP1;
  v8f acc0 = {}, acc1 = {};
#pragma unroll
  for (int k0 = 0; k0 < KP1; k0 += 32) {
    // A fragment (ISA 16-bit 16x32 layout): lanes 0-15 K{0..7,16..23}, lanes 16-31 K{8..15,24..31}
    v16h a = cat16(*(const v8h*)(arow + k0 + sel * 8),
                   *(const v8h*)(arow + k0 + 16 + sel * 8));
    // B fragments: same K order as A: lanes 0-15 K{0..7,16..23}, lanes 16-31 K{8..15,24..31}
    v16h b0 = cat16(*(const v8h*)(bc0 + k0 + sel * 8),
                    *(const v8h*)(bc0 + k0 + 16 + sel * 8));
    v16h b1 = cat16(*(const v8h*)(bc1 + k0 + sel * 8),
                    *(const v8h*)(bc1 + k0 + 16 + sel * 8));
    acc0 = __builtin_amdgcn_wmma_f32_16x16x32_f16(false, a, false, b0, (short)0, acc0, false, false);
    acc1 = __builtin_amdgcn_wmma_f32_16x16x32_f16(false, a, false, b1, (short)0, acc1, false, false);
  }
  // C/D layout: lane<16 -> col=lane, VGPR r = row r; lane>=16 -> col=lane-16, row 8+r
  float* o0 = D + (size_t)(rb + sel * 8) * C1 + cb + t;
#pragma unroll
  for (int r = 0; r < 8; ++r) o0[(size_t)r * C1] = acc0[r];
  float* o1 = o0 + 16;
#pragma unroll
  for (int r = 0; r < 8; ++r) o1[(size_t)r * C1] = acc1[r];
}

// ---------------- generic WMMA GEMM (used for layer 2, K=128, 48 cols) ----------------
template <int K, int OS>
__global__ __launch_bounds__(256) void gemm_wmma_kernel(
    const _Float16* __restrict__ A, const _Float16* __restrict__ Bt,
    float* __restrict__ D) {
  const int lane = threadIdx.x & 31;
  const int wave = threadIdx.x >> 5;
  const int t    = lane & 15;
  const int sel  = lane >> 4;
  const int rb   = blockIdx.x << 4;
  const int cb   = wave << 4;
  const _Float16* arow = A  + (size_t)(rb + t) * K;
  const _Float16* bcol = Bt + (size_t)(cb + t) * K;
  v8f acc = {};
#pragma unroll
  for (int k0 = 0; k0 < K; k0 += 32) {
    v16h a = cat16(*(const v8h*)(arow + k0 + sel * 8),
                   *(const v8h*)(arow + k0 + 16 + sel * 8));
    v16h b = cat16(*(const v8h*)(bcol + k0 + sel * 8),
                   *(const v8h*)(bcol + k0 + 16 + sel * 8));
    acc = __builtin_amdgcn_wmma_f32_16x16x32_f16(false, a, false, b, (short)0, acc, false, false);
  }
  float* orow = D + (size_t)(rb + sel * 8) * OS + cb + t;
#pragma unroll
  for (int r = 0; r < 8; ++r) orow[(size_t)r * OS] = acc[r];
}

// ---------------- attention logits per node ----------------
__global__ void alpha1_kernel(const float* __restrict__ h1,
                              const float* __restrict__ asw, const float* __restrict__ adw,
                              float* __restrict__ as1, float* __restrict__ ad1) {
  int i = blockIdx.x * blockDim.x + threadIdx.x;
  if (i >= NN * HEADS) return;
  int n = i >> 3, h = i & 7;
  const float* hp = h1 + (size_t)n * C1 + h * HID;
  const float* sw = asw + h * HID;
  const float* dw = adw + h * HID;
  float s = 0.f, d = 0.f;
#pragma unroll
  for (int c = 0; c < HID; ++c) { float v = hp[c]; s += v * sw[c]; d += v * dw[c]; }
  as1[i] = s; ad1[i] = d;
}

__global__ void alpha2_kernel(const float* __restrict__ h2,
                              const float* __restrict__ asw, const float* __restrict__ adw,
                              float* __restrict__ as2, float* __restrict__ ad2) {
  int n = blockIdx.x * blockDim.x + threadIdx.x;
  if (n >= NN) return;
  const float* hp = h2 + (size_t)n * C2P;
  float s = 0.f, d = 0.f;
#pragma unroll
  for (int c = 0; c < NCLS; ++c) { float v = hp[c]; s += v * asw[c]; d += v * adw[c]; }
  as2[n] = s; ad2[n] = d;
}

// ---------------- layer-1 edge passes ----------------
__global__ void edge_max1_kernel(const int* __restrict__ ei,
                                 const float* __restrict__ as1, const float* __restrict__ ad1,
                                 float* __restrict__ mx1) {
  int e = blockIdx.x * blockDim.x + threadIdx.x;
  if (e >= EP) return;
  int s, d; edge_sd(ei, e, s, d);
#pragma unroll
  for (int h = 0; h < HEADS; ++h) {
    float v = as1[s * HEADS + h] + ad1[d * HEADS + h];
    v = v > 0.f ? v : NEG_SLOPE * v;
    atomicMaxF(&mx1[d * HEADS + h], v);
  }
}

// One wave per edge: lane l owns channels 4l..4l+3 (head = l>>2).
// h1[src] row (512B) fetched as one coalesced 32-lane b128 load;
// atomic adds land on consecutive addresses of ag1[dst] row.
__global__ __launch_bounds__(256) void edge_agg1_kernel(
    const int* __restrict__ ei,
    const float* __restrict__ as1, const float* __restrict__ ad1,
    const float* __restrict__ mx1, const float* __restrict__ h1,
    float* __restrict__ dn1, float* __restrict__ ag1) {
  int e = (int)(((long long)blockIdx.x * blockDim.x + threadIdx.x) >> 5);  // wave = edge
  if (e >= EP) return;
  int lane = threadIdx.x & 31;
  int s, d; edge_sd(ei, e, s, d);
  int h = lane >> 2;
  float v = as1[s * HEADS + h] + ad1[d * HEADS + h];
  v = v > 0.f ? v : NEG_SLOPE * v;
  float ex = __expf(v - mx1[d * HEADS + h]);
  if ((lane & 3) == 0) atomicAdd(&dn1[d * HEADS + h], ex);
  float4 hv = *(const float4*)(h1 + (size_t)s * C1 + lane * 4);   // L2-resident gather
  float* ap = ag1 + (size_t)d * C1 + lane * 4;
  atomicAdd(ap + 0, ex * hv.x);
  atomicAdd(ap + 1, ex * hv.y);
  atomicAdd(ap + 2, ex * hv.z);
  atomicAdd(ap + 3, ex * hv.w);
}

// normalize, + bias, ELU, emit f16 activations for GEMM2
__global__ void node1_kernel(const float* __restrict__ ag1, const float* __restrict__ dn1,
                             const float* __restrict__ b1, _Float16* __restrict__ a1h) {
  long long i = (long long)blockIdx.x * blockDim.x + threadIdx.x;
  if (i >= (long long)NN * C1) return;
  int n = (int)(i >> 7), c = (int)(i & 127), h = c >> 4;
  float v = ag1[i] / (dn1[n * HEADS + h] + 1e-16f) + b1[c];
  v = v > 0.f ? v : (__expf(v) - 1.0f);              // ELU
  a1h[i] = (_Float16)v;
}

// ---------------- layer-2 edge passes ----------------
__global__ void edge_max2_kernel(const int* __restrict__ ei,
                                 const float* __restrict__ as2, const float* __restrict__ ad2,
                                 float* __restrict__ mx2) {
  int e = blockIdx.x * blockDim.x + threadIdx.x;
  if (e >= EP) return;
  int s, d; edge_sd(ei, e, s, d);
  float v = as2[s] + ad2[d];
  v = v > 0.f ? v : NEG_SLOPE * v;
  atomicMaxF(&mx2[d], v);
}

// One wave per edge; lanes 0..9 own 4 channels each (40 classes).
__global__ __launch_bounds__(256) void edge_agg2_kernel(
    const int* __restrict__ ei,
    const float* __restrict__ as2, const float* __restrict__ ad2,
    const float* __restrict__ mx2, const float* __restrict__ h2,
    float* __restrict__ dn2, float* __restrict__ ag2) {
  int e = (int)(((long long)blockIdx.x * blockDim.x + threadIdx.x) >> 5);
  if (e >= EP) return;
  int lane = threadIdx.x & 31;
  int s, d; edge_sd(ei, e, s, d);
  float v = as2[s] + ad2[d];
  v = v > 0.f ? v : NEG_SLOPE * v;
  float ex = __expf(v - mx2[d]);
  if (lane == 0) atomicAdd(&dn2[d], ex);
  if (lane < 10) {
    float4 hv = *(const float4*)(h2 + (size_t)s * C2P + lane * 4);
    float* ap = ag2 + (size_t)d * NCLS + lane * 4;
    atomicAdd(ap + 0, ex * hv.x);
    atomicAdd(ap + 1, ex * hv.y);
    atomicAdd(ap + 2, ex * hv.z);
    atomicAdd(ap + 3, ex * hv.w);
  }
}

// normalize + bias + log_softmax
__global__ void final_kernel(const float* __restrict__ ag2, const float* __restrict__ dn2,
                             const float* __restrict__ b2, float* __restrict__ out) {
  int n = blockIdx.x * blockDim.x + threadIdx.x;
  if (n >= NN) return;
  float inv = 1.0f / (dn2[n] + 1e-16f);
  float v[NCLS];
  float mx = -3.402823466e38f;
#pragma unroll
  for (int c = 0; c < NCLS; ++c) {
    v[c] = ag2[(size_t)n * NCLS + c] * inv + b2[c];
    mx = fmaxf(mx, v[c]);
  }
  float sum = 0.f;
#pragma unroll
  for (int c = 0; c < NCLS; ++c) sum += __expf(v[c] - mx);
  float ls = __logf(sum) + mx;
#pragma unroll
  for (int c = 0; c < NCLS; ++c) out[(size_t)n * NCLS + c] = v[c] - ls;
}

// ---------------- host launcher ----------------
extern "C" void kernel_launch(void* const* d_in, const int* in_sizes, int n_in,
                              void* d_out, int out_size, void* d_ws, size_t ws_size,
                              hipStream_t stream) {
  (void)in_sizes; (void)n_in; (void)out_size; (void)ws_size;
  const float* x    = (const float*)d_in[0];
  const int*   ei   = (const int*)d_in[1];
  const float* W1   = (const float*)d_in[2];
  const float* asw1 = (const float*)d_in[3];
  const float* adw1 = (const float*)d_in[4];
  const float* b1   = (const float*)d_in[5];
  const float* W2   = (const float*)d_in[6];
  const float* asw2 = (const float*)d_in[7];
  const float* adw2 = (const float*)d_in[8];
  const float* b2   = (const float*)d_in[9];
  float* out = (float*)d_out;

  char* wsp = (char*)d_ws;
  size_t off = 0;
  auto carve = [&](size_t bytes) -> void* {
    void* p = wsp + off;
    off = (off + bytes + 255) & ~(size_t)255;
    return p;
  };
  _Float16* xh  = (_Float16*)carve((size_t)NN * KP1 * 2);
  _Float16* w1t = (_Float16*)carve((size_t)C1 * KP1 * 2);
  float*    h1  = (float*)   carve((size_t)NN * C1 * 4);
  float*    as1 = (float*)   carve((size_t)NN * HEADS * 4);
  float*    ad1 = (float*)   carve((size_t)NN * HEADS * 4);
  float*    mx1 = (float*)   carve((size_t)NN * HEADS * 4);
  float*    dn1 = (float*)   carve((size_t)NN * HEADS * 4);
  float*    ag1 = (float*)   carve((size_t)NN * C1 * 4);
  _Float16* a1h = (_Float16*)carve((size_t)NN * C1 * 2);
  _Float16* w2t = (_Float16*)carve((size_t)C2P * C1 * 2);
  float*    h2  = (float*)   carve((size_t)NN * C2P * 4);
  float*    as2 = (float*)   carve((size_t)NN * 4);
  float*    ad2 = (float*)   carve((size_t)NN * 4);
  float*    mx2 = (float*)   carve((size_t)NN * 4);
  float*    dn2 = (float*)   carve((size_t)NN * 4);
  float*    ag2 = (float*)   carve((size_t)NN * NCLS * 4);

  const int T = 256;
  auto G = [](long long n, int t) { return (unsigned)((n + t - 1) / t); };

  // ---- layer 1 ----
  cvt_x_kernel <<<G((long long)NN * KP1, T), T, 0, stream>>>(x, xh);
  cvt_w1_kernel<<<G(C1 * KP1, T),           T, 0, stream>>>(W1, w1t);
  gemm1_wmma_kernel<<<NN / 16, 128, 0, stream>>>(xh, w1t, h1);  // 4 waves x 32 cols
  alpha1_kernel<<<G(NN * HEADS, T), T, 0, stream>>>(h1, asw1, adw1, as1, ad1);

  fill_u32_kernel<<<G(NN * HEADS, T), T, 0, stream>>>((unsigned*)mx1, NN * HEADS, 0xFF800000u);
  fill_u32_kernel<<<G(NN * HEADS, T), T, 0, stream>>>((unsigned*)dn1, NN * HEADS, 0u);
  fill_u32_kernel<<<G((long long)NN * C1, T), T, 0, stream>>>((unsigned*)ag1, (long long)NN * C1, 0u);

  edge_max1_kernel<<<G(EP, T), T, 0, stream>>>(ei, as1, ad1, mx1);
  edge_agg1_kernel<<<G((long long)EP * 32, T), T, 0, stream>>>(ei, as1, ad1, mx1, h1, dn1, ag1);
  node1_kernel<<<G((long long)NN * C1, T), T, 0, stream>>>(ag1, dn1, b1, a1h);

  // ---- layer 2 ----
  cvt_w2_kernel<<<G(C2P * C1, T), T, 0, stream>>>(W2, w2t);
  gemm_wmma_kernel<C1, C2P><<<NN / 16, 96, 0, stream>>>(a1h, w2t, h2);  // 3 waves x 16 cols
  alpha2_kernel<<<G(NN, T), T, 0, stream>>>(h2, asw2, adw2, as2, ad2);

  fill_u32_kernel<<<G(NN, T), T, 0, stream>>>((unsigned*)mx2, NN, 0xFF800000u);
  fill_u32_kernel<<<G(NN, T), T, 0, stream>>>((unsigned*)dn2, NN, 0u);
  fill_u32_kernel<<<G((long long)NN * NCLS, T), T, 0, stream>>>((unsigned*)ag2, (long long)NN * NCLS, 0u);

  edge_max2_kernel<<<G(EP, T), T, 0, stream>>>(ei, as2, ad2, mx2);
  edge_agg2_kernel<<<G((long long)EP * 32, T), T, 0, stream>>>(ei, as2, ad2, mx2, h2, dn2, ag2);
  final_kernel<<<G(NN, T), T, 0, stream>>>(ag2, dn2, b2, out);
}
